// Self_Attn_395136991701
// MI455X (gfx1250) — matrix operand order repaired and checked by reading, not compiled, not run on hardware
//
#include <hip/hip_runtime.h>
#include <cstdint>
#include <cstddef>

// ---------------------------------------------------------------------------
// Self-attention-over-hidden (B=8, N=2048, H=1024) for gfx1250 (CDNA5).
//   0. x -> bf16 ; W{q,k,v} -> W^T bf16
//   1. qT,kT = (x@W+b)^T (B,H,N) bf16 ; v = x@Wv+bv (B,N,H) bf16
//   2. energy[b] = qT[b]*kT[b]^T (f32)
//   3. softmax rows -> attn bf16
//   4. out[b] = v[b]*attn[b]^T ; epilogue gamma*acc + x
// GEMMs: v_wmma_f32_16x16x32_bf16, double-buffered LDS tiles fed by
// GLOBAL_LOAD_ASYNC_TO_LDS_B128 (ASYNCcnt) when the builtin is available.
// ---------------------------------------------------------------------------

typedef __attribute__((ext_vector_type(16))) __bf16 v16bf;
typedef __attribute__((ext_vector_type(8)))  __bf16 v8bf;
typedef __attribute__((ext_vector_type(4)))  __bf16 v4bf;
typedef __attribute__((ext_vector_type(8)))  float  v8f;
typedef __attribute__((ext_vector_type(4)))  int    v4i;

#define H_DIM 1024
#define SEQ_N 2048
#define BATCH 8
#define SEQ_SHIFT 11   // log2(SEQ_N)

static constexpr int BM = 128;
static constexpr int BN = 128;
static constexpr int BK = 32;
static constexpr int LDS_PAD = 8;  // row stride 40 bf16 = 80B (16B aligned)

enum GemmMode { PROJ_N = 0, PROJ_T = 1, ENERGY = 2, OUTF = 3 };

// ---- CDNA5 async global->LDS copy support ---------------------------------
#if defined(__has_builtin)
#  if __has_builtin(__builtin_amdgcn_global_load_async_to_lds_b128)
#    define HAVE_ASYNC_LDS 1
#  else
#    define HAVE_ASYNC_LDS 0
#  endif
#  if __has_builtin(__builtin_amdgcn_s_wait_asynccnt)
#    define HAVE_WAIT_ASYNC 1
#  else
#    define HAVE_WAIT_ASYNC 0
#  endif
#else
#  define HAVE_ASYNC_LDS 0
#  define HAVE_WAIT_ASYNC 0
#endif

#if HAVE_ASYNC_LDS
// prototype (from clang diagnostics): (v4i AS1*, v4i AS3*, imm, imm)
#define ASYNC_COPY16(ldsp, gp)                                                \
  __builtin_amdgcn_global_load_async_to_lds_b128(                             \
      (__attribute__((address_space(1))) v4i*)(uintptr_t)(gp),                \
      (__attribute__((address_space(3))) v4i*)(uint32_t)(uintptr_t)(ldsp),    \
      0, 0)
#if HAVE_WAIT_ASYNC
#define WAIT_ASYNC(n) __builtin_amdgcn_s_wait_asynccnt(n)
#else
#define WAIT_ASYNC(n) asm volatile("s_wait_asynccnt " #n ::: "memory")
#endif
#endif  // HAVE_ASYNC_LDS

__device__ __forceinline__ v16bf cat16(v8bf lo, v8bf hi) {
  return __builtin_shufflevector(lo, hi, 0, 1, 2, 3, 4, 5, 6, 7,
                                         8, 9, 10, 11, 12, 13, 14, 15);
}

// ---------------------------------------------------------------------------
__global__ __launch_bounds__(256) void f32_to_bf16_kernel(
    const float* __restrict__ in, __bf16* __restrict__ out, size_t n) {
  size_t i = ((size_t)blockIdx.x * blockDim.x + threadIdx.x) * 4;
  if (i + 3 < n) {
    float4 f = *(const float4*)(in + i);
    v4bf o;
    o[0] = (__bf16)f.x; o[1] = (__bf16)f.y;
    o[2] = (__bf16)f.z; o[3] = (__bf16)f.w;
    *(v4bf*)(out + i) = o;
  }
}

// ---------------------------------------------------------------------------
__global__ __launch_bounds__(256) void transpose_w_kernel(
    const float* __restrict__ Wq, const float* __restrict__ Wk,
    const float* __restrict__ Wv, __bf16* __restrict__ WqT,
    __bf16* __restrict__ WkT, __bf16* __restrict__ WvT, int H) {
  const float* W = (blockIdx.z == 0) ? Wq : ((blockIdx.z == 1) ? Wk : Wv);
  __bf16* WT = (blockIdx.z == 0) ? WqT : ((blockIdx.z == 1) ? WkT : WvT);

  __shared__ float tile[32][33];
  const int bx = blockIdx.x * 32;
  const int by = blockIdx.y * 32;
  const int tx = threadIdx.x & 31;
  const int ty = threadIdx.x >> 5;

#pragma unroll
  for (int r = ty; r < 32; r += 8)
    tile[r][tx] = W[(size_t)(by + r) * H + bx + tx];
  __syncthreads();
#pragma unroll
  for (int r = ty; r < 32; r += 8)
    WT[(size_t)(bx + r) * H + by + tx] = (__bf16)tile[tx][r];
}

// ---------------------------------------------------------------------------
// C(M,N) = A(M,K) * BT(N,K)^T, bf16 in / f32 accum. 256 thr = 8 waves (2x4),
// wave tile 64x32 = 4x2 WMMA tiles. Double-buffered LDS, async DMA staging.
// ---------------------------------------------------------------------------
template <int MODE>
__global__ __launch_bounds__(256) void gemm_bf16_wmma(
    const __bf16* __restrict__ Aall, int lda, size_t aBatch,
    const __bf16* __restrict__ BTall, int ldb, size_t bBatch,
    const float* __restrict__ bias,
    __bf16* __restrict__ outB,
    float* __restrict__ outFall, size_t oBatch,
    const float* __restrict__ xall, size_t xBatch,
    const float* __restrict__ gamma,
    int N, int K) {
  const __bf16* A  = Aall + (size_t)blockIdx.z * aBatch;
  const __bf16* BT = BTall + (size_t)blockIdx.z * bBatch;

  const int tid  = threadIdx.x;
  const int lane = tid & 31;
  const int wave = tid >> 5;
  const int wm   = wave >> 2;  // 0..1
  const int wn   = wave & 3;   // 0..3
  const int rowBase = blockIdx.y * BM;
  const int colBase = blockIdx.x * BN;

  __shared__ __align__(16) __bf16 As[2][BM][BK + LDS_PAD];
  __shared__ __align__(16) __bf16 Bs[2][BN][BK + LDS_PAD];

  v8f acc[4][2];
#pragma unroll
  for (int mt = 0; mt < 4; ++mt)
#pragma unroll
    for (int nt = 0; nt < 2; ++nt)
      acc[mt][nt] = (v8f){0.f, 0.f, 0.f, 0.f, 0.f, 0.f, 0.f, 0.f};

  // each thread owns one 16-element half-row of the A tile and the B tile
  const int lrow = tid >> 1;        // 0..127
  const int lcol = (tid & 1) * 16;  // 0 or 16
  const __bf16* aRow = A + (size_t)(rowBase + lrow) * lda + lcol;
  const __bf16* bRow = BT + (size_t)(colBase + lrow) * ldb + lcol;

  const int T = K / BK;

#if HAVE_ASYNC_LDS
#define STAGE_TILE(buf, k0)                                                   \
  do {                                                                        \
    ASYNC_COPY16(&As[buf][lrow][lcol], aRow + (k0));                          \
    ASYNC_COPY16(&As[buf][lrow][lcol + 8], aRow + (k0) + 8);                  \
    ASYNC_COPY16(&Bs[buf][lrow][lcol], bRow + (k0));                          \
    ASYNC_COPY16(&Bs[buf][lrow][lcol + 8], bRow + (k0) + 8);                  \
  } while (0)
#else
#define STAGE_TILE(buf, k0)                                                   \
  do {                                                                        \
    v8bf a0 = *(const v8bf*)(aRow + (k0));                                    \
    v8bf a1 = *(const v8bf*)(aRow + (k0) + 8);                                \
    v8bf b0 = *(const v8bf*)(bRow + (k0));                                    \
    v8bf b1 = *(const v8bf*)(bRow + (k0) + 8);                                \
    *(v8bf*)&As[buf][lrow][lcol] = a0;                                        \
    *(v8bf*)&As[buf][lrow][lcol + 8] = a1;                                    \
    *(v8bf*)&Bs[buf][lrow][lcol] = b0;                                        \
    *(v8bf*)&Bs[buf][lrow][lcol + 8] = b1;                                    \
  } while (0)
#endif

  STAGE_TILE(0, 0);

  for (int kt = 0; kt < T; ++kt) {
    const int cur = kt & 1;
    if (kt + 1 < T) {
      STAGE_TILE(cur ^ 1, (kt + 1) * BK);  // prefetch next tile into other buf
#if HAVE_ASYNC_LDS
      WAIT_ASYNC(4);  // in-order: <=4 outstanding => tile kt has landed
#endif
    } else {
#if HAVE_ASYNC_LDS
      WAIT_ASYNC(0);
#endif
    }
    __syncthreads();

    // B fragments: lane holds column n = l%16, K run (l/16)*16 + 0..15
    v16bf bfrag[2];
    {
      const int kb = (lane >> 4) * 16;
#pragma unroll
      for (int nt = 0; nt < 2; ++nt) {
        const int n = wn * 32 + nt * 16 + (lane & 15);
        v8bf lo = *(const v8bf*)&Bs[cur][n][kb];
        v8bf hi = *(const v8bf*)&Bs[cur][n][kb + 8];
        bfrag[nt] = cat16(lo, hi);
      }
    }
    // A fragments: lane holds row m = l%16, K run (l/16)*16 + 0..15
#pragma unroll
    for (int mt = 0; mt < 4; ++mt) {
      const int m  = wm * 64 + mt * 16 + (lane & 15);
      const int kb = (lane >> 4) * 16;
      v8bf lo = *(const v8bf*)&As[cur][m][kb];
      v8bf hi = *(const v8bf*)&As[cur][m][kb + 8];
      v16bf afrag = cat16(lo, hi);
#pragma unroll
      for (int nt = 0; nt < 2; ++nt)
        acc[mt][nt] = __builtin_amdgcn_wmma_f32_16x16x32_bf16(
            false, afrag, false, bfrag[nt], (short)0, acc[mt][nt], false,
            false);
    }
    __syncthreads();  // protect buf `cur` before it is re-written at kt+2
  }
#undef STAGE_TILE

  // ------------------------- epilogue / store -------------------------
  float g = 0.f;
  if (MODE == OUTF) g = gamma[0];
  float* outF = nullptr;
  const float* xres = nullptr;
  if (MODE == ENERGY || MODE == OUTF)
    outF = outFall + (size_t)blockIdx.z * oBatch;
  if (MODE == OUTF) xres = xall + (size_t)blockIdx.z * xBatch;

  const int rsel = (lane >> 4) << 3;
  const int gcolB = colBase + wn * 32 + (lane & 15);
  const int growB = rowBase + wm * 64 + rsel;

#pragma unroll
  for (int mt = 0; mt < 4; ++mt) {
#pragma unroll
    for (int nt = 0; nt < 2; ++nt) {
      const int gcol = gcolB + nt * 16;
#pragma unroll
      for (int r = 0; r < 8; ++r) {
        const int grow = growB + mt * 16 + r;
        float val = acc[mt][nt][r];
        if (MODE == PROJ_N) {
          val += bias[gcol];
          outB[(size_t)grow * N + gcol] = (__bf16)val;
        } else if (MODE == PROJ_T) {
          val += bias[gcol];
          const int b = grow >> SEQ_SHIFT;
          const int n = grow & (SEQ_N - 1);
          outB[((size_t)b * N + gcol) * SEQ_N + n] = (__bf16)val;
        } else if (MODE == ENERGY) {
          outF[(size_t)grow * N + gcol] = val;
        } else {
          outF[(size_t)grow * N + gcol] =
              g * val + xres[(size_t)grow * N + gcol];
        }
      }
    }
  }
}

// ---------------------------------------------------------------------------
__global__ __launch_bounds__(256) void softmax_rows_kernel(
    const float* __restrict__ E, __bf16* __restrict__ P, int cols) {
  const int row = blockIdx.x;
  const float* e = E + (size_t)row * cols;
  __bf16* p = P + (size_t)row * cols;
  const int t = threadIdx.x;

  float v[4];
  float mx = -3.0e38f;
#pragma unroll
  for (int i = 0; i < 4; ++i) {
    v[i] = e[t + i * 256];
    mx = fmaxf(mx, v[i]);
  }
  __shared__ float red[256];
  red[t] = mx;
  __syncthreads();
  for (int s = 128; s > 0; s >>= 1) {
    if (t < s) red[t] = fmaxf(red[t], red[t + s]);
    __syncthreads();
  }
  mx = red[0];
  __syncthreads();

  float sum = 0.f;
#pragma unroll
  for (int i = 0; i < 4; ++i) {
    v[i] = __expf(v[i] - mx);
    sum += v[i];
  }
  red[t] = sum;
  __syncthreads();
  for (int s = 128; s > 0; s >>= 1) {
    if (t < s) red[t] += red[t + s];
    __syncthreads();
  }
  const float inv = 1.0f / red[0];
#pragma unroll
  for (int i = 0; i < 4; ++i) p[t + i * 256] = (__bf16)(v[i] * inv);
}

// ---------------------------------------------------------------------------
extern "C" void kernel_launch(void* const* d_in, const int* in_sizes, int n_in,
                              void* d_out, int out_size, void* d_ws,
                              size_t ws_size, hipStream_t stream) {
  (void)in_sizes; (void)n_in; (void)out_size; (void)ws_size;
  const float* x     = (const float*)d_in[0];
  const float* Wq    = (const float*)d_in[1];
  const float* bq    = (const float*)d_in[2];
  const float* Wk    = (const float*)d_in[3];
  const float* bk    = (const float*)d_in[4];
  const float* Wv    = (const float*)d_in[5];
  const float* bv    = (const float*)d_in[6];
  const float* gamma = (const float*)d_in[7];
  float* out = (float*)d_out;

  const size_t BNH = (size_t)BATCH * SEQ_N * H_DIM;
  const size_t BHH = (size_t)BATCH * H_DIM * H_DIM;
  const size_t HH  = (size_t)H_DIM * H_DIM;

  char* ws = (char*)d_ws;
  size_t off = 0;
  auto carve = [&](size_t bytes) -> void* {
    void* p = ws + off;
    off += (bytes + 255) & ~(size_t)255;
    return p;
  };
  __bf16* xb     = (__bf16*)carve(BNH * 2);
  __bf16* WqT    = (__bf16*)carve(HH * 2);
  __bf16* WkT    = (__bf16*)carve(HH * 2);
  __bf16* WvT    = (__bf16*)carve(HH * 2);
  __bf16* qT     = (__bf16*)carve(BNH * 2);
  __bf16* kT     = (__bf16*)carve(BNH * 2);
  __bf16* vB     = (__bf16*)carve(BNH * 2);
  float*  energy = (float*)carve(BHH * 4);
  __bf16* attn   = (__bf16*)carve(BHH * 2);

  f32_to_bf16_kernel<<<(int)(BNH / 4 / 256), 256, 0, stream>>>(x, xb, BNH);
  {
    dim3 g(H_DIM / 32, H_DIM / 32, 3);
    transpose_w_kernel<<<g, 256, 0, stream>>>(Wq, Wk, Wv, WqT, WkT, WvT,
                                              H_DIM);
  }
  {
    dim3 g(H_DIM / BN, (BATCH * SEQ_N) / BM, 1);
    gemm_bf16_wmma<PROJ_T><<<g, 256, 0, stream>>>(
        xb, H_DIM, 0, WqT, H_DIM, 0, bq, qT, nullptr, 0, nullptr, 0, nullptr,
        H_DIM, H_DIM);
    gemm_bf16_wmma<PROJ_T><<<g, 256, 0, stream>>>(
        xb, H_DIM, 0, WkT, H_DIM, 0, bk, kT, nullptr, 0, nullptr, 0, nullptr,
        H_DIM, H_DIM);
    gemm_bf16_wmma<PROJ_N><<<g, 256, 0, stream>>>(
        xb, H_DIM, 0, WvT, H_DIM, 0, bv, vB, nullptr, 0, nullptr, 0, nullptr,
        H_DIM, H_DIM);
  }
  {
    dim3 g(H_DIM / BN, H_DIM / BM, BATCH);
    gemm_bf16_wmma<ENERGY><<<g, 256, 0, stream>>>(
        qT, SEQ_N, (size_t)H_DIM * SEQ_N, kT, SEQ_N, (size_t)H_DIM * SEQ_N,
        nullptr, nullptr, energy, HH, nullptr, 0, nullptr, H_DIM, SEQ_N);
  }
  softmax_rows_kernel<<<BATCH * H_DIM, 256, 0, stream>>>(energy, attn, H_DIM);
  {
    dim3 g(H_DIM / BN, SEQ_N / BM, BATCH);
    gemm_bf16_wmma<OUTF><<<g, 256, 0, stream>>>(
        vB, H_DIM, (size_t)SEQ_N * H_DIM, attn, H_DIM, HH, nullptr, nullptr,
        out, (size_t)SEQ_N * H_DIM, x, (size_t)SEQ_N * H_DIM, gamma, H_DIM,
        H_DIM);
  }
}
